// VQ_VAE_26482768347879
// MI455X (gfx1250) — matrix operand order repaired and checked by reading, not compiled, not run on hardware
//
#include <hip/hip_runtime.h>
#include <hip/hip_bf16.h>
#include <stdint.h>

// ---------------------------------------------------------------------------
// VQ-VAE forward for MI455X (gfx1250, wave32).
//  * Convs / transposed convs -> one templated implicit-GEMM kernel using
//    v_wmma_f32_16x16x32_bf16 (fp32 accumulate). Conv geometry is template
//    constants; all spatial dims are power-of-two shifts.
//  * A-gather uses branchless clamped addressing: unconditional clause-able
//    global loads + post-select, no exec-mask churn, single wait per tile.
//  * Weight panels n-major [N_pad][Kc_pad]; B tile via
//    global_load_async_to_lds_b128 (ASYNCcnt) with uint4 fallback.
//  * VQ nearest-neighbor: memory-bound 134MB codebook stream, z (4MB) resident
//    in the 192MB L2, global_prefetch_b8 ahead of the stream.
// ---------------------------------------------------------------------------

typedef __attribute__((ext_vector_type(16))) __bf16 v16bf;
typedef __attribute__((ext_vector_type(8)))  float  v8f;

#define HIDC   128
#define LATC   64
#define KCODES 512
#define BATCH  16
#define HLSZ   32
#define DDIM   (LATC * HLSZ * HLSZ)   // 65536

#if __has_builtin(__builtin_amdgcn_global_load_async_to_lds_b128)
#define HAVE_ASYNC_COPY 1
typedef int async_v4i __attribute__((vector_size(16)));
typedef __attribute__((address_space(1))) async_v4i* async_gptr;
typedef __attribute__((address_space(3))) async_v4i* async_lptr;
#else
#define HAVE_ASYNC_COPY 0
#endif

__device__ __forceinline__ unsigned short f2bf_bits(float f) {
  unsigned u = __builtin_bit_cast(unsigned, f);
  u += 0x7FFFu + ((u >> 16) & 1u);          // round-to-nearest-even
  return (unsigned short)(u >> 16);
}

__device__ __forceinline__ v8f v8f_zero() {
  v8f v;
#pragma unroll
  for (int i = 0; i < 8; ++i) v[i] = 0.0f;
  return v;
}

// ---------------------------------------------------------------------------
// Weight prep: fp32 OIHW (or transpose+flip for conv_t) -> bf16 n-major panel
// Bw[N_pad][Kc_pad], zero padded.
// ---------------------------------------------------------------------------
__global__ __launch_bounds__(256)
void prep_weights_kernel(const float* __restrict__ w, __bf16* __restrict__ Bw,
                         int d0, int d1, int kh, int kw,
                         int Cin, int Cout, int Kc_pad, int N_pad, int tflip)
{
  int idx = blockIdx.x * 256 + threadIdx.x;
  if (idx >= Kc_pad * N_pad) return;
  int co = idx / Kc_pad;          // n-major layout
  int kk = idx - co * Kc_pad;
  unsigned short v = 0;
  int khw = kh * kw;
  if (co < Cout && kk < Cin * khw) {
    int ci = kk / khw;
    int rs = kk - ci * khw;
    int r  = rs / kw;
    int s  = rs - r * kw;
    float f;
    if (tflip)  // conv_t: w[d0=Cin][d1=Cout], spatially flipped
      f = w[(((size_t)ci * d1 + co) * kh + (kh - 1 - r)) * kw + (kw - 1 - s)];
    else        // conv: w[d0=Cout][d1=Cin]
      f = w[(((size_t)co * d1 + ci) * kh + r) * kw + s];
    v = f2bf_bits(f);
  }
  Bw[idx] = __builtin_bit_cast(__bf16, v);
}

// ---------------------------------------------------------------------------
// Implicit-GEMM conv / conv_t (lhs-dilated) with bf16 WMMA.
// Block: 128 threads (4 waves). Tile: 64(M) x 64(N) x 32(K).
// Wave quadrant 32x32 = 4x v_wmma_f32_16x16x32_bf16 accumulators.
// ---------------------------------------------------------------------------
template <int KH, int KW, int STRIDE, int PAD, int DIL>
__global__ __launch_bounds__(128)
void conv_gemm_wmma(const float* __restrict__ in, const __bf16* __restrict__ Bw,
                    const float* __restrict__ bias, const float* __restrict__ residual,
                    float* __restrict__ out,
                    int Cin, int Hin, int Win,
                    int in_wshift, int in_hwshift,   // log2(Win), log2(Hin*Win)
                    int Cout,
                    int wshift, int hwshift,         // log2(Wout), log2(Hout*Wout)
                    int Kc_real, int Kc_pad, int N_pad, int relu_in)
{
  constexpr int KHW = KH * KW;

  __shared__ __align__(16) __bf16 As[64][32];   // [m][k]
  __shared__ __align__(16) __bf16 Bs[64][32];   // [n][k]

  const int tid  = threadIdx.x;
  const int lane = tid & 31;
  const int wave = tid >> 5;
  const int wm   = (wave >> 1) * 32;
  const int wn   = (wave & 1) * 32;

  const int m0 = blockIdx.x * 64;
  const int n0 = blockIdx.y * 64;

  const int wmask  = (1 << wshift) - 1;
  const int hwmask = (1 << hwshift) - 1;

  // This thread's A-gather row (fixed across the K loop)
  const int arow = tid & 63;
  const int akh  = (tid >> 6) * 16;
  const int gm   = m0 + arow;
  const int an   = gm >> hwshift;
  const int arem = gm & hwmask;
  const int aoh  = arem >> wshift;
  const int aow  = arem & wmask;
  const float* inbase = in + ((size_t)an * Cin << in_hwshift);

  // B-tile copy coords: 32B per thread, row (tid>>1), half-row (tid&1)
  const int brow = tid >> 1;
  const int bhalf = (tid & 1) * 16;

  v8f c00 = v8f_zero(), c01 = v8f_zero(), c10 = v8f_zero(), c11 = v8f_zero();

  const int frow = lane & 15;
  const int fk8  = (lane >> 4) * 8;    // A frag K-group
  const int fk16 = (lane >> 4) * 16;   // unused; B frag also uses fk8

  for (int k0 = 0; k0 < Kc_pad; k0 += 32) {
    // ---- B tile: straight copy of the n-major bf16 panel ----
    {
      const __bf16* gs = Bw + (size_t)(n0 + brow) * Kc_pad + k0 + bhalf;
      __bf16* ls = &Bs[brow][bhalf];
#if HAVE_ASYNC_COPY
      __builtin_amdgcn_global_load_async_to_lds_b128((async_gptr)gs, (async_lptr)ls, 0, 0);
      __builtin_amdgcn_global_load_async_to_lds_b128((async_gptr)gs, (async_lptr)ls, 16, 0);
#else
      const uint4* g4 = (const uint4*)gs;
      uint4* l4 = (uint4*)ls;
      l4[0] = g4[0];
      l4[1] = g4[1];
#endif
    }

    // ---- gather A tile: branchless clamped addressing, unconditional loads,
    //      post-select. All 16 loads can issue in one clause. ----
    {
      float vals[16];
      bool  preds[16];
#pragma unroll
      for (int j = 0; j < 16; ++j) {
        int kk = k0 + akh + j;
        bool pred = (kk < Kc_real);
        int kks = pred ? kk : 0;                 // clamp so ci stays in range
        int ci = kks / KHW;                      // compile-time divisor
        int rs = kks - ci * KHW;
        int r  = rs / KW;
        int s  = rs - r * KW;
        int ihd = aoh * STRIDE - PAD + r;
        int iwd = aow * STRIDE - PAD + s;
        bool ok = (ihd >= 0) && (iwd >= 0);
        if (DIL != 1) ok = ok && ((ihd % DIL) == 0) && ((iwd % DIL) == 0);
        int ih = (DIL == 1) ? ihd : (ihd / DIL);
        int iw = (DIL == 1) ? iwd : (iwd / DIL);
        ok = ok && (ih < Hin) && (iw < Win);
        pred = pred && ok;
        int ihs = pred ? ih : 0;                 // clamp to a safe address
        int iws = pred ? iw : 0;
        vals[j]  = inbase[((size_t)ci << in_hwshift) + (ihs << in_wshift) + iws];
        preds[j] = pred;
      }
      unsigned pk[8];
#pragma unroll
      for (int jp = 0; jp < 8; ++jp) {
        unsigned short h2[2];
#pragma unroll
        for (int e = 0; e < 2; ++e) {
          int j = jp * 2 + e;
          float v = preds[j] ? vals[j] : 0.0f;
          if (relu_in) v = fmaxf(v, 0.0f);
          h2[e] = f2bf_bits(v);
        }
        pk[jp] = (unsigned)h2[0] | ((unsigned)h2[1] << 16);
      }
      uint4* dst = (uint4*)&As[arow][akh];
      dst[0] = make_uint4(pk[0], pk[1], pk[2], pk[3]);
      dst[1] = make_uint4(pk[4], pk[5], pk[6], pk[7]);
    }

#if HAVE_ASYNC_COPY
#if __has_builtin(__builtin_amdgcn_s_wait_asynccnt)
    __builtin_amdgcn_s_wait_asynccnt(0);
#else
    asm volatile("s_wait_asynccnt 0" ::: "memory");
#endif
#endif
    __syncthreads();

    // ---- fragments per ISA VGPR layouts ----
    v16bf a0, a1, b0, b1;
#pragma unroll
    for (int h = 0; h < 8; ++h) {
      a0[h]     = As[wm + frow][fk8 + h];
      a0[h + 8] = As[wm + frow][16 + fk8 + h];
      a1[h]     = As[wm + 16 + frow][fk8 + h];
      a1[h + 8] = As[wm + 16 + frow][16 + fk8 + h];
    }
#pragma unroll
    for (int j = 0; j < 16; ++j) {
      b0[j] = Bs[wn + frow][(j & 8) * 2 + fk8 + (j & 7)];
      b1[j] = Bs[wn + 16 + frow][(j & 8) * 2 + fk8 + (j & 7)];
    }

    c00 = __builtin_amdgcn_wmma_f32_16x16x32_bf16(false, a0, false, b0, (short)0, c00, false, false);
    c01 = __builtin_amdgcn_wmma_f32_16x16x32_bf16(false, a0, false, b1, (short)0, c01, false, false);
    c10 = __builtin_amdgcn_wmma_f32_16x16x32_bf16(false, a1, false, b0, (short)0, c10, false, false);
    c11 = __builtin_amdgcn_wmma_f32_16x16x32_bf16(false, a1, false, b1, (short)0, c11, false, false);
    __syncthreads();
  }

  // ---- epilogue: bias (+ optional residual), scatter to NCHW fp32 ----
  const int ecol = lane & 15;
  const int erow = (lane >> 4) * 8;
  v8f accs[2][2] = {{c00, c01}, {c10, c11}};
#pragma unroll
  for (int mt = 0; mt < 2; ++mt) {
#pragma unroll
    for (int nt = 0; nt < 2; ++nt) {
      int gnn = n0 + wn + nt * 16 + ecol;
      if (gnn >= Cout) continue;
      float bv = bias[gnn];
#pragma unroll
      for (int v = 0; v < 8; ++v) {
        int gmm = m0 + wm + mt * 16 + erow + v;
        int on  = gmm >> hwshift;
        int orr = gmm & hwmask;
        int ooh = orr >> wshift;
        int oow = orr & wmask;
        size_t oidx = (((size_t)on * Cout + gnn) << hwshift) + (ooh << wshift) + oow;
        float val = accs[mt][nt][v] + bv;
        if (residual) val += residual[oidx];
        out[oidx] = val;
      }
    }
  }
}

// ---------------------------------------------------------------------------
// VQ kernels
// ---------------------------------------------------------------------------
__global__ __launch_bounds__(256)
void row_sq_sum_kernel(const float* __restrict__ M, float* __restrict__ outv, int ncols)
{
  __shared__ float red[256];
  const int tid = threadIdx.x;
  const float* row = M + (size_t)blockIdx.x * ncols;
  float s = 0.0f;
  for (int j = tid; j < ncols; j += 256) { float v = row[j]; s += v * v; }
  red[tid] = s;
  __syncthreads();
  for (int o = 128; o > 0; o >>= 1) {
    if (tid < o) red[tid] += red[tid + o];
    __syncthreads();
  }
  if (tid == 0) outv[blockIdx.x] = red[0];
}

// One block per code k: stream W[k][:] once (HBM), z stays in L2.
__global__ __launch_bounds__(256)
void vq_dist_kernel(const float* __restrict__ Z, const float* __restrict__ W,
                    const float* __restrict__ zz, const float* __restrict__ ww,
                    float* __restrict__ dist, int D, int K)
{
  const int k = blockIdx.x;
  const int tid = threadIdx.x;
  const float* wr = W + (size_t)k * D;
  float acc[16];
#pragma unroll
  for (int b = 0; b < 16; ++b) acc[b] = 0.0f;
  for (int j = tid; j < D; j += 256) {
    __builtin_prefetch(wr + j + 4096, 0, 1);   // global_prefetch_b8 on the W stream
    float wv = wr[j];
#pragma unroll
    for (int b = 0; b < 16; ++b) acc[b] += wv * Z[(size_t)b * D + j];
  }
  __shared__ float red[16][256];
#pragma unroll
  for (int b = 0; b < 16; ++b) red[b][tid] = acc[b];
  __syncthreads();
  for (int o = 128; o > 0; o >>= 1) {
    if (tid < o) {
#pragma unroll
      for (int b = 0; b < 16; ++b) red[b][tid] += red[b][tid + o];
    }
    __syncthreads();
  }
  if (tid < 16) dist[(size_t)tid * K + k] = zz[tid] + ww[k] - 2.0f * red[tid][0];
}

__global__ __launch_bounds__(256)
void vq_argmin_kernel(const float* __restrict__ dist, int* __restrict__ idx, int K)
{
  const int b = blockIdx.x;
  const int tid = threadIdx.x;
  float best = 3.4e38f;
  int   bi   = 0x3FFFFFFF;
  for (int k = tid; k < K; k += 256) {
    float d = dist[(size_t)b * K + k];
    if (d < best || (d == best && k < bi)) { best = d; bi = k; }
  }
  __shared__ float bv[256];
  __shared__ int   bix[256];
  bv[tid] = best; bix[tid] = bi;
  __syncthreads();
  for (int o = 128; o > 0; o >>= 1) {
    if (tid < o) {
      if (bv[tid + o] < bv[tid] || (bv[tid + o] == bv[tid] && bix[tid + o] < bix[tid])) {
        bv[tid] = bv[tid + o]; bix[tid] = bix[tid + o];
      }
    }
    __syncthreads();
  }
  if (tid == 0) idx[b] = bix[0];
}

// Gather z_q, emit z_e/z_q output copies, build latent with the reference's
// reshape(B,H,W,C)->NCHW quirk: latent[b,c,h,w] = z_q[b][h*2048 + w*64 + c].
__global__ __launch_bounds__(256)
void vq_outputs_kernel(const float* __restrict__ Z, const float* __restrict__ W,
                       const int* __restrict__ idx,
                       float* __restrict__ ze1, float* __restrict__ zq1,
                       float* __restrict__ ze2, float* __restrict__ zq2,
                       float* __restrict__ latent)
{
  size_t t = (size_t)blockIdx.x * 256 + threadIdx.x;
  if (t >= (size_t)BATCH * DDIM) return;
  int b = (int)(t >> 16);
  int j = (int)(t & 65535);
  float ze = Z[t];
  float zq = W[(size_t)idx[b] * DDIM + j];
  ze1[t] = ze; ze2[t] = ze;
  zq1[t] = zq; zq2[t] = zq;
  int c = j & 63;
  int w = (j >> 6) & 31;
  int h = j >> 11;
  latent[(((size_t)b * LATC + c) * HLSZ + h) * HLSZ + w] = zq;
}

// ---------------------------------------------------------------------------
// Host orchestration
// ---------------------------------------------------------------------------
#define LAUNCH_CONV(KH, KW, ST, PD, DL, in, Bw, bias, res, outp, Cin, Hin, Win, Cout, Hout, Wout, Kr, Kcp, Np, relu) \
  do {                                                                                             \
    dim3 g((unsigned)((BATCH * (Hout) * (Wout)) / 64), (unsigned)((Np) / 64));                     \
    conv_gemm_wmma<KH, KW, ST, PD, DL><<<g, 128, 0, stream>>>(                                     \
        in, Bw, bias, res, outp, Cin, Hin, Win,                                                    \
        __builtin_ctz((unsigned)(Win)), __builtin_ctz((unsigned)((Hin) * (Win))),                  \
        Cout,                                                                                      \
        __builtin_ctz((unsigned)(Wout)), __builtin_ctz((unsigned)((Hout) * (Wout))),               \
        Kr, Kcp, Np, relu);                                                                        \
  } while (0)

extern "C" void kernel_launch(void* const* d_in, const int* in_sizes, int n_in,
                              void* d_out, int out_size, void* d_ws, size_t ws_size,
                              hipStream_t stream) {
  (void)in_sizes; (void)n_in; (void)out_size; (void)ws_size;

  const float* x     = (const float*)d_in[0];
  const float* Wcode = (const float*)d_in[1];
  const float* p[30];
  for (int i = 0; i < 30; ++i) p[i] = (const float*)d_in[i];
  // enc: 2..15 = conv1_w,b conv2_w,b b1w1,b1 b1w2,b2 b2w1,b1 b2w2,b2 out_w,b
  // dec: 16..29 = in_w,b b1w1,b1 b1w2,b2 b2w1,b1 b2w2,b2 ct1_w,b ct2_w,b

  char* base = (char*)d_ws;
  size_t off = 0;
  auto alloc = [&](size_t bytes) -> void* {
    off = (off + 255) & ~(size_t)255;
    void* ptr = base + off;
    off += bytes;
    return ptr;
  };

  // bf16 weight panels, n-major [N_pad][Kc_pad]
  __bf16* w_c1   = (__bf16*)alloc((size_t)128 * 64   * 2);
  __bf16* w_c2   = (__bf16*)alloc((size_t)128 * 2048 * 2);
  __bf16* w_eb1a = (__bf16*)alloc((size_t)128 * 1152 * 2);
  __bf16* w_eb1b = (__bf16*)alloc((size_t)128 * 128  * 2);
  __bf16* w_eb2a = (__bf16*)alloc((size_t)128 * 1152 * 2);
  __bf16* w_eb2b = (__bf16*)alloc((size_t)128 * 128  * 2);
  __bf16* w_eout = (__bf16*)alloc((size_t)64  * 128  * 2);
  __bf16* w_din  = (__bf16*)alloc((size_t)128 * 64   * 2);
  __bf16* w_db1a = (__bf16*)alloc((size_t)128 * 1152 * 2);
  __bf16* w_db1b = (__bf16*)alloc((size_t)128 * 128  * 2);
  __bf16* w_db2a = (__bf16*)alloc((size_t)128 * 1152 * 2);
  __bf16* w_db2b = (__bf16*)alloc((size_t)128 * 128  * 2);
  __bf16* w_ct1  = (__bf16*)alloc((size_t)128 * 2048 * 2);
  __bf16* w_ct2  = (__bf16*)alloc((size_t)64  * 2048 * 2);

  // activation buffers
  float* bufA1 = (float*)alloc((size_t)BATCH * HIDC * 64 * 64 * 4); // 32MB (also D4)
  float* P0    = (float*)alloc((size_t)BATCH * HIDC * 32 * 32 * 4);
  float* P1    = (float*)alloc((size_t)BATCH * HIDC * 32 * 32 * 4);
  float* P2    = (float*)alloc((size_t)BATCH * HIDC * 32 * 32 * 4);
  float* Zbuf  = (float*)alloc((size_t)BATCH * DDIM * 4);
  float* Lbuf  = (float*)alloc((size_t)BATCH * DDIM * 4);
  float* zzb   = (float*)alloc(16 * 4);
  float* wwb   = (float*)alloc(KCODES * 4);
  float* distb = (float*)alloc((size_t)16 * KCODES * 4);
  int*   idxb  = (int*)alloc(16 * 4);

  float* recon = (float*)d_out;                 // (16,3,128,128)
  float* ze1   = recon + 786432;
  float* zq1   = ze1 + (size_t)BATCH * DDIM;
  float* ze2   = zq1 + (size_t)BATCH * DDIM;
  float* zq2   = ze2 + (size_t)BATCH * DDIM;

  auto prep = [&](const float* w, __bf16* Bw, int d0, int d1, int kh, int kw,
                  int Cin, int Cout, int Kcp, int Np, int tflip) {
    int total = Kcp * Np;
    prep_weights_kernel<<<(total + 255) / 256, 256, 0, stream>>>(
        w, Bw, d0, d1, kh, kw, Cin, Cout, Kcp, Np, tflip);
  };

  // ---- weight prep ----
  prep(p[2],  w_c1,   128, 3,   4, 4, 3,   128, 64,   128, 0);
  prep(p[4],  w_c2,   128, 128, 4, 4, 128, 128, 2048, 128, 0);
  prep(p[6],  w_eb1a, 128, 128, 3, 3, 128, 128, 1152, 128, 0);
  prep(p[8],  w_eb1b, 128, 128, 1, 1, 128, 128, 128,  128, 0);
  prep(p[10], w_eb2a, 128, 128, 3, 3, 128, 128, 1152, 128, 0);
  prep(p[12], w_eb2b, 128, 128, 1, 1, 128, 128, 128,  128, 0);
  prep(p[14], w_eout, 64,  128, 1, 1, 128, 64,  128,  64,  0);
  prep(p[16], w_din,  64,  128, 1, 1, 64,  128, 64,   128, 1);   // conv_t 1x1
  prep(p[18], w_db1a, 128, 128, 3, 3, 128, 128, 1152, 128, 0);
  prep(p[20], w_db1b, 128, 128, 1, 1, 128, 128, 128,  128, 0);
  prep(p[22], w_db2a, 128, 128, 3, 3, 128, 128, 1152, 128, 0);
  prep(p[24], w_db2b, 128, 128, 1, 1, 128, 128, 128,  128, 0);
  prep(p[26], w_ct1,  128, 128, 4, 4, 128, 128, 2048, 128, 1);   // conv_t
  prep(p[28], w_ct2,  128, 3,   4, 4, 128, 3,   2048, 64,  1);   // conv_t

  // ---- encoder ----
  LAUNCH_CONV(4,4,2,1,1, x,     w_c1,   p[3],  nullptr, bufA1, 3,   128, 128, 128, 64, 64, 48,   64,   128, 0);
  LAUNCH_CONV(4,4,2,1,1, bufA1, w_c2,   p[5],  nullptr, P0,    128, 64,  64,  128, 32, 32, 2048, 2048, 128, 0);
  LAUNCH_CONV(3,3,1,1,1, P0,    w_eb1a, p[7],  nullptr, P1,    128, 32,  32,  128, 32, 32, 1152, 1152, 128, 1);
  LAUNCH_CONV(1,1,1,0,1, P1,    w_eb1b, p[9],  P0,      P2,    128, 32,  32,  128, 32, 32, 128,  128,  128, 1);
  LAUNCH_CONV(3,3,1,1,1, P2,    w_eb2a, p[11], nullptr, P1,    128, 32,  32,  128, 32, 32, 1152, 1152, 128, 1);
  LAUNCH_CONV(1,1,1,0,1, P1,    w_eb2b, p[13], P2,      P0,    128, 32,  32,  128, 32, 32, 128,  128,  128, 1);
  LAUNCH_CONV(1,1,1,0,1, P0,    w_eout, p[15], nullptr, Zbuf,  128, 32,  32,  64,  32, 32, 128,  128,  64,  0);

  // ---- vector quantization ----
  row_sq_sum_kernel<<<KCODES, 256, 0, stream>>>(Wcode, wwb, DDIM);
  row_sq_sum_kernel<<<16, 256, 0, stream>>>(Zbuf, zzb, DDIM);
  vq_dist_kernel<<<KCODES, 256, 0, stream>>>(Zbuf, Wcode, zzb, wwb, distb, DDIM, KCODES);
  vq_argmin_kernel<<<16, 256, 0, stream>>>(distb, idxb, KCODES);
  vq_outputs_kernel<<<(BATCH * DDIM) / 256, 256, 0, stream>>>(
      Zbuf, Wcode, idxb, ze1, zq1, ze2, zq2, Lbuf);

  // ---- decoder ----
  LAUNCH_CONV(1,1,1,0,1, Lbuf,  w_din,  p[17], nullptr, P2,    64,  32, 32, 128, 32,  32,  64,   64,   128, 0);
  LAUNCH_CONV(3,3,1,1,1, P2,    w_db1a, p[19], nullptr, P1,    128, 32, 32, 128, 32,  32,  1152, 1152, 128, 1);
  LAUNCH_CONV(1,1,1,0,1, P1,    w_db1b, p[21], P2,      P0,    128, 32, 32, 128, 32,  32,  128,  128,  128, 1);
  LAUNCH_CONV(3,3,1,1,1, P0,    w_db2a, p[23], nullptr, P1,    128, 32, 32, 128, 32,  32,  1152, 1152, 128, 1);
  LAUNCH_CONV(1,1,1,0,1, P1,    w_db2b, p[25], P0,      P2,    128, 32, 32, 128, 32,  32,  128,  128,  128, 1);
  // ct1: lhs-dilated conv, dil=2, pad = k-1-pad = 2, stride(conv)=1
  LAUNCH_CONV(4,4,1,2,2, P2,    w_ct1,  p[27], nullptr, bufA1, 128, 32, 32, 128, 64,  64,  2048, 2048, 128, 0);
  // ct2: -> reconstruct (N padded 3 -> 64, only co<3 written)
  LAUNCH_CONV(4,4,1,2,2, bufA1, w_ct2,  p[29], nullptr, recon, 128, 64, 64, 3,   128, 128, 2048, 2048, 64,  0);
}
